// PatchPonderPAM_1451698946854
// MI455X (gfx1250) — matrix operand order repaired and checked by reading, not compiled, not run on hardware
//
#include <hip/hip_runtime.h>
#include <hip/hip_bf16.h>

typedef _Float16 half_t;
typedef __attribute__((ext_vector_type(16))) _Float16 v16h;
typedef __attribute__((ext_vector_type(8)))  _Float16 v8h;
typedef __attribute__((ext_vector_type(8)))  float    v8f;

__device__ __forceinline__ float silu_f(float x) { return x / (1.f + __expf(-x)); }

__device__ __forceinline__ v16h load_frag(const half_t* __restrict__ p) {
    v8h x0 = *(const v8h*)p;
    v8h x1 = *(const v8h*)(p + 8);
    v16h r;
#pragma unroll
    for (int i = 0; i < 8; ++i) { r[i] = x0[i]; r[8 + i] = x1[i]; }
    return r;
}

// ---------------------------------------------------------------------------
// Weight pre-swizzle: GEMM B fragments.  One thread writes one lane's 32B
// fragment record:  elem e -> W[(k32*32 + (e<8 ? hi*8+e : 16+hi*8+e-8))*N + nt*16 + lm]
// ---------------------------------------------------------------------------
__global__ void pack_w_gemm(const float* __restrict__ W, half_t* __restrict__ out,
                            int N, int Kd)
{
    const int k32s = Kd >> 5;
    const int total = (N >> 4) * k32s * 32;
    int i = blockIdx.x * blockDim.x + threadIdx.x;
    if (i >= total) return;
    const int lane = i & 31;
    const int k32  = (i >> 5) % k32s;
    const int nt   = (i >> 5) / k32s;
    const int lm = lane & 15, hi = lane >> 4;
    half_t* dst = out + (size_t)i * 16;
#pragma unroll
    for (int e = 0; e < 16; ++e) {
        int K = k32 * 32 + ((e < 8) ? (hi * 8 + e) : (16 + hi * 8 + (e - 8)));
        dst[e] = (half_t)W[(size_t)K * N + nt * 16 + lm];
    }
}

// ---------------------------------------------------------------------------
// Weight pre-swizzle: conv A fragments (K-order = tap r outer, ci inner).
// ---------------------------------------------------------------------------
__global__ void pack_w_conv(const float* __restrict__ wgt, half_t* __restrict__ out,
                            int Co, int Ci)
{
    const int nch = (Ci + 31) >> 5;
    const int total = (Co >> 4) * 9 * nch * 32;
    int i = blockIdx.x * blockDim.x + threadIdx.x;
    if (i >= total) return;
    const int lane = i & 31;
    int rest = i >> 5;
    const int ch  = rest % nch; rest /= nch;
    const int r   = rest % 9;
    const int cot = rest / 9;
    const int lm = lane & 15, hi = lane >> 4;
    half_t* dst = out + (size_t)i * 16;
#pragma unroll
    for (int e = 0; e < 16; ++e) {
        int cl = (e < 8) ? (hi * 8 + e) : (16 + hi * 8 + (e - 8));
        int ci = ch * 32 + cl;
        float v = (ci < Ci) ? wgt[((size_t)(cot * 16 + lm) * Ci + ci) * 9 + r] : 0.f;
        dst[e] = (half_t)v;
    }
}

// ---------------------------------------------------------------------------
// Convert 3 hint channels (f32, batch stride 21ch) into a dense f16 stem input.
// ---------------------------------------------------------------------------
__global__ void cvt_hint(const float* __restrict__ hint, half_t* __restrict__ out,
                         int chan_off, int total)   // total = 4*3*512*512
{
    int i = blockIdx.x * blockDim.x + threadIdx.x;
    if (i >= total) return;
    int hw = i & (512 * 512 - 1);
    int rest = i >> 18;
    int c = rest % 3;
    int b = rest / 3;
    out[i] = (half_t)hint[((size_t)b * 21 + chan_off + c) * (512 * 512) + hw];
}

// ---------------------------------------------------------------------------
// Implicit-GEMM conv 3x3 SAME + bias + SiLU, f16 in / f16-or-f32 out.
// One wave computes a 32(Co) x 16(pixel) tile (2 co-tiles share one gather).
// ---------------------------------------------------------------------------
template<int CI>
__global__ void conv_silu_wmma(const half_t* __restrict__ in,   // Bn x CI x Hi x Wi (f16)
                               const half_t* __restrict__ wp,   // packed A fragments
                               const float* __restrict__ bias,  // Co
                               half_t* __restrict__ outH,       // one of outH/outF non-null
                               float*  __restrict__ outF,
                               int Bn, int Hi, int Wi,
                               int Co, int Ho, int Wo, int stride)
{
    constexpr int NCH = (CI + 31) / 32;
    const int wave = (blockIdx.x * blockDim.x + threadIdx.x) >> 5;
    const int lane = threadIdx.x & 31;
    const int npix   = Bn * Ho * Wo;
    const int ptiles = npix >> 4;
    const int cog    = (Co >> 4) >> 1;           // co-tile pairs
    if (wave >= ptiles * cog) return;            // wave-uniform: EXEC stays all-ones
    const int cp = wave / ptiles;
    const int pt = wave - cp * ptiles;
    const int cot0 = cp * 2, cot1 = cot0 + 1;
    const int pb = pt << 4;

    const int lm = lane & 15, hi = lane >> 4;
    const int p    = pb + lm;
    const int HoWo = Ho * Wo;
    const int b    = p / HoWo;
    const int rem  = p - b * HoWo;
    const int oy   = rem / Wo;
    const int ox   = rem - oy * Wo;
    const int iy0  = oy * stride - 1;
    const int ix0  = ox * stride - 1;
    const size_t HiWi = (size_t)Hi * Wi;
    const half_t* __restrict__ inb = in + (size_t)b * CI * HiWi;

    v8f acc0 = {}, acc1 = {};
#pragma unroll
    for (int r = 0; r < 9; ++r) {                // tap (ky,kx) — constants after unroll
        const int ky = r / 3, kx = r - ky * 3;
        const int iy = iy0 + ky, ix = ix0 + kx;
        const bool ok = ((unsigned)iy < (unsigned)Hi) & ((unsigned)ix < (unsigned)Wi);
        const half_t* __restrict__ ibase =
            inb + (size_t)(ok ? iy : 0) * Wi + (ok ? ix : 0);   // always-valid address
#pragma unroll
        for (int ch = 0; ch < NCH; ++ch) {
            // ---- B fragment: f16 activations, col = pixel, K-local = ci ----
            v16h bf;
#pragma unroll
            for (int i = 0; i < 16; ++i) {
                int ci  = ch * 32 + ((i < 8) ? (hi * 8 + i) : (16 + hi * 8 + (i - 8)));
                int cic = (ci < CI) ? ci : 0;                   // folds (CI constexpr)
                half_t v = ibase[(size_t)cic * HiWi];
                bf[i] = (ok && (ci < CI)) ? v : (half_t)0.f;
            }
            // ---- A fragments: packed weights, two co-tiles, coalesced ----
            const size_t rec = ((size_t)(r * NCH + ch) * 32 + lane) * 16;
            v16h a0 = load_frag(wp + ((size_t)cot0 * 9 * NCH * 32 * 16) + rec);
            v16h a1 = load_frag(wp + ((size_t)cot1 * 9 * NCH * 32 * 16) + rec);
            acc0 = __builtin_amdgcn_wmma_f32_16x16x32_f16(false, a0, false, bf,
                                                          (short)0, acc0, false, false);
            acc1 = __builtin_amdgcn_wmma_f32_16x16x32_f16(false, a1, false, bf,
                                                          (short)0, acc1, false, false);
        }
    }

    // epilogue: D layout — VGPR r, lanes 0-15 -> M=r, lanes 16-31 -> M=r+8
#pragma unroll
    for (int r = 0; r < 8; ++r) {
        int co0 = cot0 * 16 + (hi ? 8 : 0) + r;
        int co1 = cot1 * 16 + (hi ? 8 : 0) + r;
        float v0 = silu_f(acc0[r] + bias[co0]);
        float v1 = silu_f(acc1[r] + bias[co1]);
        size_t i0 = ((size_t)b * Co + co0) * HoWo + rem;
        size_t i1 = ((size_t)b * Co + co1) * HoWo + rem;
        if (outH) { outH[i0] = (half_t)v0; outH[i1] = (half_t)v1; }
        if (outF) { outF[i0] = v0;         outF[i1] = v1; }
    }
}

// ---------------------------------------------------------------------------
// Generic GEMM:  out(MxN) = A_f16(MxK) @ W(KxN) + bias (+resid) (+SiLU)
// W pre-swizzled to f16 fragment records, staged LDS-resident via a per-wave
// double-buffered global_load_async_to_lds_b128 pipeline (ASYNCcnt tracked).
// One wave computes 16 x 64 (A fragment reused 4x).  Block = 4 waves.
// Grid: x = mtiles/4, y = N-tile group (block-uniform).
// ---------------------------------------------------------------------------
template<int NT>
__global__ void __launch_bounds__(128)
gemm_wmma(const half_t* __restrict__ A,     // M x Kd row-major
          const half_t* __restrict__ Bp,    // packed fragments
          const float*  __restrict__ bias,  // N
          const float*  __restrict__ resid, // M x N or nullptr
          float*  __restrict__ outF,        // may be nullptr
          half_t* __restrict__ outH,        // may be nullptr
          int M, int N, int Kd, int act)
{
    static_assert(NT == 4, "async wait immediates assume NT==4");
    __shared__ half_t sB[4][2][NT * 512];          // 4 waves x 2 bufs x 4KB
    const int w    = threadIdx.x >> 5;
    const int lane = threadIdx.x & 31;
    const int grp  = blockIdx.y;
    const int mtile = blockIdx.x * 4 + w;
    const int k32s = Kd >> 5;
    if ((mtile << 4) >= M) return;                 // wave-uniform guard
    const int mb = mtile << 4;

    const int lm = lane & 15, hi = lane >> 4;
    const half_t* __restrict__ arow = A + (size_t)(mb + lm) * Kd + (hi ? 8 : 0);

    // async-stage one 32-chunk of packed B fragments (NT records) into LDS buf
    auto stageB = [&](int k32, int buf) {
#pragma unroll
        for (int t = 0; t < NT; ++t) {
            const half_t* src = Bp + (((size_t)(grp * NT + t) * k32s + k32) * 32 + lane) * 16;
            unsigned dst = (unsigned)(size_t)&sB[w][buf][(t * 32 + lane) * 16];
            asm volatile("global_load_async_to_lds_b128 %0, %1, off"
                         :: "v"(dst), "v"(src) : "memory");
            asm volatile("global_load_async_to_lds_b128 %0, %1, off offset:16"
                         :: "v"(dst), "v"(src) : "memory");
        }
    };

    v8f acc[NT];
#pragma unroll
    for (int t = 0; t < NT; ++t) acc[t] = (v8f){};

    stageB(0, 0);
    for (int k32 = 0; k32 < k32s; ++k32) {
        const int buf = k32 & 1;
        if (k32 + 1 < k32s) {
            stageB(k32 + 1, buf ^ 1);
            asm volatile("s_wait_asynccnt 0x8" ::: "memory");   // 2*NT just issued
        } else {
            asm volatile("s_wait_asynccnt 0x0" ::: "memory");
        }
        __builtin_prefetch(arow + k32 * 32 + 64, 0, 1);         // global_prefetch_b8
        v16h a;
        {
            v8h a0 = *(const v8h*)(arow + k32 * 32);
            v8h a1 = *(const v8h*)(arow + k32 * 32 + 16);
#pragma unroll
            for (int i = 0; i < 8; ++i) { a[i] = a0[i]; a[8 + i] = a1[i]; }
        }
#pragma unroll
        for (int t = 0; t < NT; ++t) {
            const int rb = (t * 32 + lane) * 16;
            v8h b0 = *(const v8h*)&sB[w][buf][rb];
            v8h b1 = *(const v8h*)&sB[w][buf][rb + 8];
            v16h bf;
#pragma unroll
            for (int i = 0; i < 8; ++i) { bf[i] = b0[i]; bf[8 + i] = b1[i]; }
            acc[t] = __builtin_amdgcn_wmma_f32_16x16x32_f16(false, a, false, bf,
                                                            (short)0, acc[t], false, false);
        }
    }

#pragma unroll
    for (int t = 0; t < NT; ++t) {
        const int n = (grp * NT + t) * 16 + lm;
#pragma unroll
        for (int r = 0; r < 8; ++r) {
            int row = mb + (hi ? 8 : 0) + r;
            size_t idx = (size_t)row * N + n;
            float v = acc[t][r] + bias[n];
            if (resid) v += resid[idx];
            if (act == 1) v = silu_f(v);
            if (outF) outF[idx] = v;
            if (outH) outH[idx] = (half_t)v;
        }
    }
}

// ---------------------------------------------------------------------------
// Context token:  ctx = timeMLP(temb) + txtMLP(mean(txt,1)); ctx_c = ctx@c_w+c_b
// ---------------------------------------------------------------------------
__global__ void ctx_kernel(const float* __restrict__ temb, const float* __restrict__ txt,
                           const float* __restrict__ tw1, const float* __restrict__ tb1,
                           const float* __restrict__ tw2, const float* __restrict__ tb2,
                           const float* __restrict__ xw1, const float* __restrict__ xb1,
                           const float* __restrict__ xw2, const float* __restrict__ xb2,
                           const float* __restrict__ cw,  const float* __restrict__ cb,
                           float* __restrict__ ctx_c)     // 4 x 192
{
    __shared__ float txm[768];
    __shared__ float h1[192];
    __shared__ float h2[192];
    __shared__ float ctx[192];
    const int b = blockIdx.x;
    const int t = threadIdx.x;           // 256 threads

    for (int d = t; d < 768; d += 256) {
        float s = 0.f;
        const float* base = txt + ((size_t)b * 77) * 768 + d;
        for (int j = 0; j < 77; ++j) s += base[(size_t)j * 768];
        txm[d] = s * (1.f / 77.f);
    }
    __syncthreads();
    for (int d = t; d < 192; d += 256) {
        const float* te = temb + (size_t)b * 320;
        float s1 = tb1[d];
        for (int j = 0; j < 320; ++j) s1 += te[j] * tw1[(size_t)j * 192 + d];
        h1[d] = silu_f(s1);
        float s2 = xb1[d];
        for (int j = 0; j < 768; ++j) s2 += txm[j] * xw1[(size_t)j * 192 + d];
        h2[d] = silu_f(s2);
    }
    __syncthreads();
    for (int d = t; d < 192; d += 256) {
        float s1 = tb2[d], s2 = xb2[d];
        for (int j = 0; j < 192; ++j) {
            s1 += h1[j] * tw2[(size_t)j * 192 + d];
            s2 += h2[j] * xw2[(size_t)j * 192 + d];
        }
        ctx[d] = s1 + s2;
    }
    __syncthreads();
    for (int d = t; d < 192; d += 256) {
        float s = cb[d];
        for (int j = 0; j < 192; ++j) s += ctx[j] * cw[(size_t)j * 192 + d];
        ctx_c[b * 192 + d] = s;
    }
}

// ---------------------------------------------------------------------------
// Gather per-pixel tokens (feats (k,b,c,h,w) f32 -> tokens f16 (pixel*3+k, c))
// ---------------------------------------------------------------------------
__global__ void gather_tokens(const float* __restrict__ feats, half_t* __restrict__ tok, int total)
{
    int i = blockIdx.x * blockDim.x + threadIdx.x;
    if (i >= total) return;
    int c   = i % 192;
    int row = i / 192;                 // pixel*3 + k
    int k   = row % 3;
    int pix = row / 3;
    int b   = pix >> 12;               // / 4096
    int hw  = pix & 4095;
    float v = feats[((((size_t)k * 4 + b) * 192) + c) * 4096 + hw];
    tok[(size_t)row * 192 + c] = (half_t)v;
}

// ---------------------------------------------------------------------------
// LayerNorm, one wave per 192-wide row, wave32 shuffle reduction; f16 out.
// ---------------------------------------------------------------------------
__global__ void ln_kernel(const float* __restrict__ X, const float* __restrict__ g,
                          const float* __restrict__ bt, half_t* __restrict__ Y, int M)
{
    const int wave = (blockIdx.x * blockDim.x + threadIdx.x) >> 5;
    const int lane = threadIdx.x & 31;
    if (wave >= M) return;
    const float* x = X + (size_t)wave * 192;
    float vals[6], s = 0.f, s2 = 0.f;
#pragma unroll
    for (int i = 0; i < 6; ++i) {
        float v = x[lane + 32 * i];
        vals[i] = v; s += v; s2 += v * v;
    }
#pragma unroll
    for (int o = 16; o > 0; o >>= 1) {
        s  += __shfl_xor(s,  o, 32);
        s2 += __shfl_xor(s2, o, 32);
    }
    const float mean = s * (1.f / 192.f);
    const float var  = s2 * (1.f / 192.f) - mean * mean;
    const float inv  = rsqrtf(var + 1e-5f);
    half_t* y = Y + (size_t)wave * 192;
#pragma unroll
    for (int i = 0; i < 6; ++i) {
        int d = lane + 32 * i;
        y[d] = (half_t)((vals[i] - mean) * inv * g[d] + bt[d]);
    }
}

// ---------------------------------------------------------------------------
// Tri-attention: per (pixel, head): K=3 tokens, dh=24, context-modulated K/V.
// ---------------------------------------------------------------------------
__global__ void attn_kernel(const half_t* __restrict__ Q, const half_t* __restrict__ Kx,
                            const half_t* __restrict__ V, const float* __restrict__ ctx_c,
                            half_t* __restrict__ O, int npix)
{
    int i = blockIdx.x * blockDim.x + threadIdx.x;
    if (i >= npix * 8) return;
    const int h   = i & 7;
    const int pix = i >> 3;
    const int b   = pix >> 12;

    float c24[24];
    const float* c = ctx_c + b * 192 + h * 24;
#pragma unroll
    for (int d = 0; d < 24; ++d) c24[d] = c[d];

    float q[3][24];
    size_t rbase = (size_t)pix * 3 * 192 + h * 24;
#pragma unroll
    for (int tk = 0; tk < 3; ++tk)
#pragma unroll
        for (int d = 0; d < 24; ++d) q[tk][d] = (float)Q[rbase + (size_t)tk * 192 + d];

    const float scale = 0.2041241452319315f;   // 1/sqrt(24)
    float probs[3][3];
#pragma unroll
    for (int a = 0; a < 3; ++a) {
        float sc[3], mx = -1e30f;
#pragma unroll
        for (int j = 0; j < 3; ++j) {
            const half_t* kr = Kx + rbase + (size_t)j * 192;
            float s = 0.f;
#pragma unroll
            for (int d = 0; d < 24; ++d) s += q[a][d] * ((float)kr[d] * c24[d]);
            sc[j] = s * scale;
            mx = fmaxf(mx, sc[j]);
        }
        float den = 0.f;
#pragma unroll
        for (int j = 0; j < 3; ++j) { sc[j] = __expf(sc[j] - mx); den += sc[j]; }
        float r = 1.f / den;
#pragma unroll
        for (int j = 0; j < 3; ++j) probs[a][j] = sc[j] * r;
    }
#pragma unroll
    for (int tk = 0; tk < 3; ++tk) {
        half_t* orow = O + rbase + (size_t)tk * 192;
#pragma unroll
        for (int d = 0; d < 24; ++d) {
            float vc0 = (float)V[rbase + d]       * c24[d];
            float vc1 = (float)V[rbase + 192 + d] * c24[d];
            float vc2 = (float)V[rbase + 384 + d] * c24[d];
            orow[d] = (half_t)(probs[tk][0] * vc0 + probs[tk][1] * vc1 + probs[tk][2] * vc2);
        }
    }
}

// ---------------------------------------------------------------------------
// Gate logits (192 -> 1 per token) + hard argmax (first-max wins, like jnp).
// ---------------------------------------------------------------------------
__global__ void gate_kernel(const float* __restrict__ T, const float* __restrict__ gw,
                            const float* __restrict__ gb, int* __restrict__ kmax, int npix)
{
    int p = blockIdx.x * blockDim.x + threadIdx.x;
    if (p >= npix) return;
    float l[3];
#pragma unroll
    for (int tk = 0; tk < 3; ++tk) {
        const float* x = T + ((size_t)p * 3 + tk) * 192;
        float s = gb[0];
        for (int j = 0; j < 192; ++j) s += x[j] * gw[j];
        l[tk] = s;
    }
    int kb = 0; float best = l[0];
    if (l[1] > best) { best = l[1]; kb = 1; }
    if (l[2] > best) { kb = 2; }
    kmax[p] = kb;
}

__global__ void select_kernel(const float* __restrict__ feats, const int* __restrict__ kmax,
                              float* __restrict__ out, int total)
{
    int i = blockIdx.x * blockDim.x + threadIdx.x;
    if (i >= total) return;
    int hw = i & 4095;
    int bc = i >> 12;
    int c  = bc % 192;
    int b  = bc / 192;
    int k  = kmax[(b << 12) | hw];
    out[i] = feats[((((size_t)k * 4 + b) * 192) + c) * 4096 + hw];
}

// ---------------------------------------------------------------------------
// Host-side orchestration
// ---------------------------------------------------------------------------
extern "C" void kernel_launch(void* const* d_in, const int* in_sizes, int n_in,
                              void* d_out, int out_size, void* d_ws, size_t ws_size,
                              hipStream_t stream)
{
    (void)in_sizes; (void)n_in; (void)out_size; (void)ws_size;
    const float* hint = (const float*)d_in[0];
    const float* temb = (const float*)d_in[1];
    const float* txt  = (const float*)d_in[2];
    // params flattened in dict insertion order; each stem = 12 tensors (c1_w..c6_b)
    const int STEM0 = 3;
    const int P_TW1 = 39, P_TB1 = 40, P_TW2 = 41, P_TB2 = 42;
    const int P_XW1 = 43, P_XB1 = 44, P_XW2 = 45, P_XB2 = 46;
    const int P_TOKW = 47, P_TOKB = 48, P_LN1G = 49, P_LN1B = 50;
    const int P_QW = 51, P_QB = 52, P_KW = 53, P_KB = 54, P_VW = 55, P_VB = 56;
    const int P_CW = 57, P_CB = 58, P_OW = 59, P_OB = 60;
    const int P_LN2G = 61, P_LN2B = 62;
    const int P_FW1 = 63, P_FB1 = 64, P_FW2 = 65, P_FB2 = 66;
    const int P_GW = 67, P_GB = 68;

    char* ws = (char*)d_ws;
    const size_t FEATS_OFF = 0;                         // 3 * 4*192*4096 f32
    const size_t FEATS_SZ  = (size_t)3 * 4 * 192 * 4096 * 4;
    const size_t PINGA_OFF = FEATS_OFF + FEATS_SZ;      // activations now f16; keep f32-size regions
    const size_t PINGA_SZ  = (size_t)4 * 32 * 512 * 512 * 4;
    const size_t PINGB_OFF = PINGA_OFF + PINGA_SZ;
    const size_t PINGB_SZ  = (size_t)4 * 64 * 256 * 256 * 4;
    const size_t FF1_OFF   = PINGB_OFF + PINGB_SZ;      //  75,497,472 B (f16)
    const size_t FF1_SZ    = (size_t)49152 * 768 * 2;
    const size_t CTXC_OFF  = FF1_OFF + FF1_SZ;          //  4*192 f32
    const size_t KMAX_OFF  = CTXC_OFF + 4096;           //  16384 int
    const size_t WPC_OFF   = KMAX_OFF + 65536;          //  packed conv weights (<=442 KB)
    const size_t WPT_OFF   = WPC_OFF + 524288;          //  5x 73,728 B (192x192 packs)
    const size_t WPQ_OFF   = WPT_OFF + 73728;
    const size_t WPK_OFF   = WPQ_OFF + 73728;
    const size_t WPV_OFF   = WPK_OFF + 73728;
    const size_t WPO_OFF   = WPV_OFF + 73728;
    const size_t WPF1_OFF  = WPO_OFF + 73728;           //  294,912 B
    const size_t WPF2_OFF  = WPF1_OFF + 294912;         //  294,912 B
    const size_t HINT16_OFF= WPF2_OFF + 294912;         //  4*3*512*512 f16 = 6,291,456 B

    half_t* pingA  = (half_t*)(ws + PINGA_OFF);
    half_t* pingB  = (half_t*)(ws + PINGB_OFF);
    float*  feats  = (float*)(ws + FEATS_OFF);
    float*  ctx_c  = (float*)(ws + CTXC_OFF);
    int*    kmax   = (int*)  (ws + KMAX_OFF);
    half_t* ff1_16 = (half_t*)(ws + FF1_OFF);
    half_t* wpc    = (half_t*)(ws + WPC_OFF);
    half_t* wp_tok = (half_t*)(ws + WPT_OFF);
    half_t* wp_q   = (half_t*)(ws + WPQ_OFF);
    half_t* wp_k   = (half_t*)(ws + WPK_OFF);
    half_t* wp_v   = (half_t*)(ws + WPV_OFF);
    half_t* wp_o   = (half_t*)(ws + WPO_OFF);
    half_t* wp_f1  = (half_t*)(ws + WPF1_OFF);
    half_t* wp_f2  = (half_t*)(ws + WPF2_OFF);
    half_t* hint16 = (half_t*)(ws + HINT16_OFF);
    // transformer buffers overlayed on pingA / pingB (stems are finished first)
    const size_t TOKSZ = (size_t)49152 * 192 * 2;       // 18,874,368
    const size_t TSZ   = (size_t)49152 * 192 * 4;       // 37,748,736
    half_t* tok16  = (half_t*)((char*)pingA);
    float*  T      = (float*) ((char*)pingA + TOKSZ);
    half_t* tn16   = (half_t*)((char*)pingA + TOKSZ + TSZ);
    half_t* attn16 = (half_t*)((char*)pingA + 2 * TOKSZ + TSZ);
    half_t* h16    = (half_t*)((char*)pingA + 3 * TOKSZ + TSZ);
    half_t* q16    = (half_t*)((char*)pingB);
    half_t* k16    = (half_t*)((char*)pingB + TOKSZ);
    half_t* v16    = (half_t*)((char*)pingB + 2 * TOKSZ);

    const int Bn = 4;
    // ---------------- stems ----------------
    struct L { int Ci, Hi, Wi, Co, Ho, Wo, s; };
    const L ls[6] = {
        {3,   512, 512, 32,  512, 512, 1},
        {32,  512, 512, 64,  256, 256, 2},
        {64,  256, 256, 64,  256, 256, 1},
        {64,  256, 256, 128, 128, 128, 2},
        {128, 128, 128, 128, 128, 128, 1},
        {128, 128, 128, 192, 64,  64,  2},
    };
    const int chan_off[3] = {0, 15, 18};
    for (int st = 0; st < 3; ++st) {
        const int pbase = STEM0 + st * 12;
        float* feat_st = feats + (size_t)st * 4 * 192 * 4096;
        // dense f16 copy of this stem's 3 hint channels
        {
            int total = 4 * 3 * 512 * 512;
            cvt_hint<<<(total + 255) / 256, 256, 0, stream>>>(hint, hint16, chan_off[st], total);
        }
        // ping-pong: L0 hint16->A, L1 A->B, L2 B->A, L3 A->B, L4 B->A, L5 A->feats(f32)
        const half_t* srcs[6] = { hint16, pingA, pingB, pingA, pingB, pingA };
        half_t*       dstH[6] = { pingA, pingB, pingA, pingB, pingA, nullptr };
        float*        dstF[6] = { nullptr, nullptr, nullptr, nullptr, nullptr, feat_st };
        for (int li = 0; li < 6; ++li) {
            const L& l = ls[li];
            const float* w  = (const float*)d_in[pbase + 2 * li];
            const float* bs = (const float*)d_in[pbase + 2 * li + 1];
            // pre-swizzle this layer's weights into WMMA A-fragment records
            {
                int nch = (l.Ci + 31) >> 5;
                int total = (l.Co >> 4) * 9 * nch * 32;
                pack_w_conv<<<(total + 255) / 256, 256, 0, stream>>>(w, wpc, l.Co, l.Ci);
            }
            long waves = ((long)Bn * l.Ho * l.Wo / 16) * ((l.Co / 16) / 2);
            int blocks = (int)((waves + 7) / 8);
            switch (l.Ci) {
            case 3:
                conv_silu_wmma<3><<<blocks, 256, 0, stream>>>(
                    srcs[li], wpc, bs, dstH[li], dstF[li], Bn, l.Hi, l.Wi, l.Co, l.Ho, l.Wo, l.s);
                break;
            case 32:
                conv_silu_wmma<32><<<blocks, 256, 0, stream>>>(
                    srcs[li], wpc, bs, dstH[li], dstF[li], Bn, l.Hi, l.Wi, l.Co, l.Ho, l.Wo, l.s);
                break;
            case 64:
                conv_silu_wmma<64><<<blocks, 256, 0, stream>>>(
                    srcs[li], wpc, bs, dstH[li], dstF[li], Bn, l.Hi, l.Wi, l.Co, l.Ho, l.Wo, l.s);
                break;
            default:
                conv_silu_wmma<128><<<blocks, 256, 0, stream>>>(
                    srcs[li], wpc, bs, dstH[li], dstF[li], Bn, l.Hi, l.Wi, l.Co, l.Ho, l.Wo, l.s);
                break;
            }
        }
    }

    // ---------------- pre-swizzle transformer weights ----------------
    {
        auto packw = [&](const float* W, half_t* dst, int N, int Kd) {
            int total = (N >> 4) * (Kd >> 5) * 32;
            pack_w_gemm<<<(total + 255) / 256, 256, 0, stream>>>(W, dst, N, Kd);
        };
        packw((const float*)d_in[P_TOKW], wp_tok, 192, 192);
        packw((const float*)d_in[P_QW],   wp_q,   192, 192);
        packw((const float*)d_in[P_KW],   wp_k,   192, 192);
        packw((const float*)d_in[P_VW],   wp_v,   192, 192);
        packw((const float*)d_in[P_OW],   wp_o,   192, 192);
        packw((const float*)d_in[P_FW1],  wp_f1,  768, 192);
        packw((const float*)d_in[P_FW2],  wp_f2,  192, 768);
    }

    // ---------------- context ----------------
    ctx_kernel<<<4, 256, 0, stream>>>(
        temb, txt,
        (const float*)d_in[P_TW1], (const float*)d_in[P_TB1],
        (const float*)d_in[P_TW2], (const float*)d_in[P_TB2],
        (const float*)d_in[P_XW1], (const float*)d_in[P_XB1],
        (const float*)d_in[P_XW2], (const float*)d_in[P_XB2],
        (const float*)d_in[P_CW],  (const float*)d_in[P_CB], ctx_c);

    // ---------------- per-pixel transformer ----------------
    const int M = 49152;                 // 16384 pixels * 3 tokens
    const int npix = 16384;
    {
        int total = M * 192;
        gather_tokens<<<(total + 255) / 256, 256, 0, stream>>>(feats, tok16, total);
    }
    auto launch_gemm = [&](const half_t* A, const half_t* Bp, const float* bias,
                           const float* resid, float* outF, half_t* outH, int N, int Kd, int act) {
        dim3 grid((M / 16) / 4, (N >> 4) / 4);
        gemm_wmma<4><<<grid, 128, 0, stream>>>(A, Bp, bias, resid, outF, outH, M, N, Kd, act);
    };
    // T = tokens @ tok_w + tok_b
    launch_gemm(tok16, wp_tok, (const float*)d_in[P_TOKB], nullptr, T, nullptr, 192, 192, 0);
    // LN1
    ln_kernel<<<(M + 7) / 8, 256, 0, stream>>>(
        T, (const float*)d_in[P_LN1G], (const float*)d_in[P_LN1B], tn16, M);
    // Q, K, V
    launch_gemm(tn16, wp_q, (const float*)d_in[P_QB], nullptr, nullptr, q16, 192, 192, 0);
    launch_gemm(tn16, wp_k, (const float*)d_in[P_KB], nullptr, nullptr, k16, 192, 192, 0);
    launch_gemm(tn16, wp_v, (const float*)d_in[P_VB], nullptr, nullptr, v16, 192, 192, 0);
    // attention
    attn_kernel<<<(npix * 8 + 255) / 256, 256, 0, stream>>>(q16, k16, v16, ctx_c, attn16, npix);
    // T += attn @ o_w + o_b   (residual, in-place)
    launch_gemm(attn16, wp_o, (const float*)d_in[P_OB], T, T, nullptr, 192, 192, 0);
    // LN2
    ln_kernel<<<(M + 7) / 8, 256, 0, stream>>>(
        T, (const float*)d_in[P_LN2G], (const float*)d_in[P_LN2B], h16, M);
    // FF1 (192->768, SiLU) to f16
    launch_gemm(h16, wp_f1, (const float*)d_in[P_FB1], nullptr, nullptr, ff1_16, 768, 192, 1);
    // T += FF1 @ ff_w2 + ff_b2 (768->192, residual, in-place)
    launch_gemm(ff1_16, wp_f2, (const float*)d_in[P_FB2], T, T, nullptr, 192, 768, 0);
    // gate + hard argmax
    gate_kernel<<<(npix + 255) / 256, 256, 0, stream>>>(
        T, (const float*)d_in[P_GW], (const float*)d_in[P_GB], kmax, npix);
    // hard select into d_out (B,192,64,64)
    {
        int total = 4 * 192 * 4096;
        select_kernel<<<(total + 255) / 256, 256, 0, stream>>>(feats, kmax, (float*)d_out, total);
    }
}
